// DiffRankNet_10754598109730
// MI455X (gfx1250) — matrix operand order repaired and checked by reading, not compiled, not run on hardware
//
#include <hip/hip_runtime.h>
#include <hip/hip_bf16.h>
#include <math.h>
#include <stdint.h>

// Problem constants (from reference)
#define N_NODES  4096
#define C_DIM    1024
#define HID_DIM  512
#define FEAT_DIM 128
#define NEDGE    (3 * N_NODES)   // concat of k=5,10,15 hyperedge sets

typedef __attribute__((ext_vector_type(16))) __bf16 v16bf;
typedef __attribute__((ext_vector_type(8)))  __bf16 v8bf;
typedef __attribute__((ext_vector_type(8)))  float  v8f;

struct Frag16 { union { v16bf v; v8bf h[2]; }; };

__device__ __forceinline__ void split_f32(float a, __bf16& hi, __bf16& lo) {
  __bf16 h = (__bf16)a;
  hi = h;
  lo = (__bf16)(a - (float)h);
}

// s_wait_asynccnt: builtin if this toolchain declares it, else inline asm.
#if defined(__has_builtin) && __has_builtin(__builtin_amdgcn_s_wait_asynccnt)
#define WAIT_ASYNC(n) __builtin_amdgcn_s_wait_asynccnt(n)
#else
#define WAIT_ASYNC(n) asm volatile("s_wait_asynccnt %0" ::"i"(n) : "memory")
#endif

// Async global->LDS copy of 16 bytes per lane (ASYNCcnt-tracked).
// ldsoff = low 32 bits of the generic shared-memory pointer (flat aperture
// mapping keeps the LDS byte offset in addr[31:0]).
__device__ __forceinline__ void async_g2l_b128(unsigned ldsoff, const void* gptr) {
  unsigned long long ga = (unsigned long long)(uintptr_t)gptr;
  asm volatile("global_load_async_to_lds_b128 %0, %1, off" ::"v"(ldsoff), "v"(ga)
               : "memory");
}

#define LDS_STRIDE 40  // 32 k-values + 8 pad -> conflict-free ds_load_b128 frags

// ---------------------------------------------------------------------------
// Split-bf16 WMMA GEMM:  C[M,Nn] = A[M,K] @ B[K,Nn], B given transposed
// (BT row-major [Nn,K]).  A,B each as (hi,lo) bf16 pair; fp32 accumulate via
// 3x v_wmma_f32_16x16x32_bf16 per K-step (drop lo*lo term, ~2^-16 rel err).
// Block = 256 thr = 8 waves; block tile 128(M) x 64(N); each wave computes a
// 16x64 strip (4 N-tiles, A fragment register-reused 4x).
// B tile (shared by all 8 waves) is staged to LDS once per block with
// double-buffered GLOBAL_LOAD_ASYNC_TO_LDS_B128 (copy of chunk i+1 overlaps
// the WMMAs of chunk i); fragments then come from ds_load_b128.
// MODE 1 fuses the cdist epilogue: D = max(n_i + n_j - 2*dot, 0).
// ---------------------------------------------------------------------------
template <int MODE>
__global__ __launch_bounds__(256)
void gemm_wmma_bf16x2(const __bf16* __restrict__ Ah, const __bf16* __restrict__ Al,
                      const __bf16* __restrict__ Bh, const __bf16* __restrict__ Bl,
                      float* __restrict__ Cout, const float* __restrict__ norms,
                      int M, int Nn, int K) {
  const int tid  = threadIdx.x;
  const int wave = tid >> 5;
  const int lane = tid & 31;
  const int r    = lane & 15;      // row-in-tile (A) / col-in-tile (B,C)
  const int g    = lane >> 4;      // lane-group (K-half select)
  const int mi   = blockIdx.y * 128 + wave * 16;
  const int nj   = blockIdx.x * 64;
  (void)M;

  __shared__ __bf16 sB[2][2][64 * LDS_STRIDE];  // [buf][hi/lo][col*stride+k]

  // Per-thread staging slice: 16B of one column's K-chunk.
  const int scol  = tid >> 2;                     // 0..63  (output column)
  const int spart = tid & 3;                      // 0..3   (8-elem sub-chunk)
  const size_t sgoff   = (size_t)(nj + scol) * K + (size_t)spart * 8;
  const unsigned sloff = (unsigned)((scol * LDS_STRIDE + spart * 8) * sizeof(__bf16));
  const unsigned ldsH0 = (unsigned)(uintptr_t)(&sB[0][0][0]) + sloff;
  const unsigned ldsL0 = (unsigned)(uintptr_t)(&sB[0][1][0]) + sloff;
  const unsigned ldsH1 = (unsigned)(uintptr_t)(&sB[1][0][0]) + sloff;
  const unsigned ldsL1 = (unsigned)(uintptr_t)(&sB[1][1][0]) + sloff;

  v8f acc[4] = {};

  // A fragment: lane holds row mi+r, K chunks {k0+8g .. +7} and {k0+16+8g .. +7}
  const __bf16* pah = Ah + (size_t)(mi + r) * K + 8 * g;
  const __bf16* pal = Al + (size_t)(mi + r) * K + 8 * g;

  const int nsteps = K / 32;

  // Prologue: stage chunk 0 into buffer 0.
  async_g2l_b128(ldsH0, Bh + sgoff);
  async_g2l_b128(ldsL0, Bl + sgoff);

  for (int i = 0; i < nsteps; ++i) {
    const int k0 = i * 32;
    if (i + 1 < nsteps) {  // stage next chunk into the other buffer
      const size_t go = sgoff + (size_t)(k0 + 32);
      if ((i & 1) == 0) {
        async_g2l_b128(ldsH1, Bh + go);
        async_g2l_b128(ldsL1, Bl + go);
      } else {
        async_g2l_b128(ldsH0, Bh + go);
        async_g2l_b128(ldsL0, Bl + go);
      }
      WAIT_ASYNC(2);  // in-order per wave: chunk i has landed
    } else {
      WAIT_ASYNC(0);
    }
    __syncthreads();  // make all waves' staged data visible

    __builtin_prefetch(pah + k0 + 128, 0, 1);  // -> global_prefetch_b8
    __builtin_prefetch(pal + k0 + 128, 0, 1);

    Frag16 ah, al;
    ah.h[0] = *(const v8bf*)(pah + k0);
    ah.h[1] = *(const v8bf*)(pah + k0 + 16);
    al.h[0] = *(const v8bf*)(pal + k0);
    al.h[1] = *(const v8bf*)(pal + k0 + 16);

    const __bf16* sbh = &sB[i & 1][0][0];
    const __bf16* sbl = &sB[i & 1][1][0];

#pragma unroll
    for (int s = 0; s < 4; ++s) {
      // B fragment from LDS: lane holds col 16s+r, K chunks {8g..8g+7} and {16+8g..16+8g+7}
      const int lo = (s * 16 + r) * LDS_STRIDE + 8 * g;
      Frag16 bh, bl;
      bh.h[0] = *(const v8bf*)(sbh + lo);
      bh.h[1] = *(const v8bf*)(sbh + lo + 16);
      bl.h[0] = *(const v8bf*)(sbl + lo);
      bl.h[1] = *(const v8bf*)(sbl + lo + 16);

      acc[s] = __builtin_amdgcn_wmma_f32_16x16x32_bf16(false, al.v, false, bh.v,
                                                       (short)0, acc[s], false, false);
      acc[s] = __builtin_amdgcn_wmma_f32_16x16x32_bf16(false, ah.v, false, bl.v,
                                                       (short)0, acc[s], false, false);
      acc[s] = __builtin_amdgcn_wmma_f32_16x16x32_bf16(false, ah.v, false, bh.v,
                                                       (short)0, acc[s], false, false);
    }
    __syncthreads();  // buffer i&1 fully consumed before it is re-staged
  }

  // C/D layout: elem e of v8f -> row = mi + 8g + e, col = nj + 16s + r
#pragma unroll
  for (int s = 0; s < 4; ++s) {
    const int col = nj + s * 16 + r;
#pragma unroll
    for (int e = 0; e < 8; ++e) {
      const int row = mi + 8 * g + e;
      float d = acc[s][e];
      if (MODE == 1) d = fmaxf(norms[row] + norms[col] - 2.0f * d, 0.0f);
      Cout[(size_t)row * Nn + col] = d;
    }
  }
}

// Split fp32 rows into (hi,lo) bf16 and compute row squared-norms.
__global__ __launch_bounds__(256)
void split_norm_kernel(const float* __restrict__ X, __bf16* __restrict__ Xhi,
                       __bf16* __restrict__ Xlo, float* __restrict__ norms) {
  const int row = blockIdx.x;
  const size_t base = (size_t)row * C_DIM;
  float s = 0.f;
  for (int c = threadIdx.x; c < C_DIM; c += 256) {
    float a = X[base + c];
    __bf16 h, l; split_f32(a, h, l);
    Xhi[base + c] = h; Xlo[base + c] = l;
    s += a * a;
  }
  __shared__ float red[256];
  red[threadIdx.x] = s; __syncthreads();
  for (int t = 128; t > 0; t >>= 1) {
    if (threadIdx.x < t) red[threadIdx.x] += red[threadIdx.x + t];
    __syncthreads();
  }
  if (threadIdx.x == 0) norms[row] = red[0];
}

// W [K,Nn] fp32 -> transposed split-bf16 pair T[Nn,K]
__global__ void wsplit_kernel(const float* __restrict__ W, __bf16* __restrict__ Thi,
                              __bf16* __restrict__ Tlo, int K, int Nn) {
  size_t i = (size_t)blockIdx.x * blockDim.x + threadIdx.x;
  if (i >= (size_t)K * Nn) return;
  int n = (int)(i / K), k = (int)(i % K);
  float a = W[(size_t)k * Nn + n];
  __bf16 h, l; split_f32(a, h, l);
  Thi[i] = h; Tlo[i] = l;
}

// Exact top-15 smallest (value asc, index asc) per row via u64 key extraction.
// d >= 0 (clamped) so float bits are order-monotone.
__global__ __launch_bounds__(256)
void topk15_kernel(const float* __restrict__ D, int* __restrict__ idx) {
  const int row = blockIdx.x;
  const float* dr = D + (size_t)row * N_NODES;
  __shared__ unsigned long long red[256];
  __shared__ unsigned long long lastSh;
  unsigned long long last = 0ULL;
  for (int t = 0; t < 15; ++t) {
    unsigned long long best = ~0ULL;
    for (int j = threadIdx.x; j < N_NODES; j += 256) {
      unsigned long long key =
          ((unsigned long long)__float_as_uint(dr[j]) << 32) | (unsigned)j;
      if ((t == 0 || key > last) && key < best) best = key;
    }
    red[threadIdx.x] = best; __syncthreads();
    for (int s = 128; s > 0; s >>= 1) {
      if (threadIdx.x < s) {
        unsigned long long o = red[threadIdx.x + s];
        if (o < red[threadIdx.x]) red[threadIdx.x] = o;
      }
      __syncthreads();
    }
    if (threadIdx.x == 0) {
      lastSh = red[0];
      idx[row * 15 + t] = (int)(red[0] & 0xffffffffULL);
    }
    __syncthreads();
    last = lastSh;
    __syncthreads();
  }
}

__device__ __forceinline__ int edge_k(int seg) { return seg == 0 ? 5 : (seg == 1 ? 10 : 15); }

__global__ void degcount_kernel(const int* __restrict__ idx, int* __restrict__ Dv) {
  int e = blockIdx.x * blockDim.x + threadIdx.x;
  if (e >= NEDGE) return;
  int seg = e >> 12, node = e & (N_NODES - 1);
  int k = edge_k(seg);
  for (int j = 0; j < k; ++j) atomicAdd(&Dv[idx[node * 15 + j]], 1);
}

__global__ void dvinv_kernel(const int* __restrict__ Dv, float* __restrict__ dvinv) {
  int v = blockIdx.x * blockDim.x + threadIdx.x;
  if (v >= N_NODES) return;
  int d = Dv[v];
  dvinv[v] = d > 0 ? rsqrtf((float)d) : 0.f;
}

// Z[e,:] = de * sum_j dvinv[idx_e[j]] * Y[idx_e[j],:]
__global__ __launch_bounds__(128)
void gather_kernel(const int* __restrict__ idx, const float* __restrict__ dvinv,
                   const float* __restrict__ Y, float* __restrict__ Z, int F) {
  int e = blockIdx.x;
  int seg = e >> 12, node = e & (N_NODES - 1);
  int k = edge_k(seg);
  float de = 1.0f / (float)k;
  __shared__ int rs[16]; __shared__ float wsm[16];
  if (threadIdx.x < k) {
    int v = idx[node * 15 + threadIdx.x];
    rs[threadIdx.x] = v; wsm[threadIdx.x] = dvinv[v];
  }
  __syncthreads();
  for (int f = threadIdx.x; f < F; f += 128) {
    float s = 0.f;
    for (int j = 0; j < k; ++j) s += wsm[j] * Y[(size_t)rs[j] * F + f];
    Z[(size_t)e * F + f] = de * s;
  }
}

// Out[v,:] += Z[e,:] for each member v of edge e  (H @ Z)
__global__ __launch_bounds__(128)
void scatter_kernel(const int* __restrict__ idx, const float* __restrict__ Z,
                    float* __restrict__ Out, int F) {
  int e = blockIdx.x;
  int seg = e >> 12, node = e & (N_NODES - 1);
  int k = edge_k(seg);
  __shared__ int rs[16];
  if (threadIdx.x < k) rs[threadIdx.x] = idx[node * 15 + threadIdx.x];
  __syncthreads();
  for (int f = threadIdx.x; f < F; f += 128) {
    float z = Z[(size_t)e * F + f];
    for (int j = 0; j < k; ++j) atomicAdd(&Out[(size_t)rs[j] * F + f], z);
  }
}

// h = relu(dvinv[v] * AG[v,:]) -> split bf16 operand for next GEMM
__global__ void scale_relu_split_kernel(const float* __restrict__ AG,
                                        const float* __restrict__ dvinv,
                                        __bf16* __restrict__ Hhi, __bf16* __restrict__ Hlo,
                                        int F) {
  size_t i = (size_t)blockIdx.x * blockDim.x + threadIdx.x;
  if (i >= (size_t)N_NODES * F) return;
  int v = (int)(i / F);
  float x = fmaxf(dvinv[v] * AG[i], 0.f);
  __bf16 h, l; split_f32(x, h, l);
  Hhi[i] = h; Hlo[i] = l;
}

// mean_f = (1/N) * sum_v dvinv[v] * AG[v,f]
__global__ __launch_bounds__(256)
void colmean_kernel(const float* __restrict__ AG, const float* __restrict__ dvinv,
                    float* __restrict__ meanOut, int F) {
  int f = blockIdx.x;
  float s = 0.f;
  for (int v = threadIdx.x; v < N_NODES; v += 256)
    s += dvinv[v] * AG[(size_t)v * F + f];
  __shared__ float red[256];
  red[threadIdx.x] = s; __syncthreads();
  for (int t = 128; t > 0; t >>= 1) {
    if (threadIdx.x < t) red[threadIdx.x] += red[threadIdx.x + t];
    __syncthreads();
  }
  if (threadIdx.x == 0) meanOut[f] = red[0] / (float)N_NODES;
}

__global__ __launch_bounds__(128)
void finalize_kernel(const float* __restrict__ mgt, const float* __restrict__ m1,
                     float* __restrict__ out) {
  __shared__ float ra[128], rb[128];
  ra[threadIdx.x] = mgt[threadIdx.x];
  rb[threadIdx.x] = m1[threadIdx.x];
  __syncthreads();
  for (int s = 64; s > 0; s >>= 1) {
    if (threadIdx.x < s) {
      ra[threadIdx.x] += ra[threadIdx.x + s];
      rb[threadIdx.x] += rb[threadIdx.x + s];
    }
    __syncthreads();
  }
  if (threadIdx.x == 0) {
    float y0 = ra[0], y1 = rb[0];
    out[0] = 1.f / (1.f + expf(-(y0 - y1)));   // sigmoid(y_compare)
    out[1] = y0;
    out[2] = y1;
  }
}

extern "C" void kernel_launch(void* const* d_in, const int* in_sizes, int n_in,
                              void* d_out, int out_size, void* d_ws, size_t ws_size,
                              hipStream_t stream) {
  (void)in_sizes; (void)n_in; (void)out_size;
  const float* fts_gt = (const float*)d_in[0];
  const float* fts1   = (const float*)d_in[1];
  const float* W1     = (const float*)d_in[2];
  const float* W2     = (const float*)d_in[3];
  float* out = (float*)d_out;

  size_t off = 0;
  auto carve = [&](size_t bytes) -> char* {
    char* p = (char*)d_ws + off;
    off += (bytes + 255) & ~(size_t)255;
    return p;
  };
  __bf16* XHI   = (__bf16*)carve((size_t)N_NODES * C_DIM * 2);
  __bf16* XLO   = (__bf16*)carve((size_t)N_NODES * C_DIM * 2);
  float*  DMAT  = (float*)carve((size_t)N_NODES * N_NODES * 4);   // reused as Z after top-k
  float*  NORMS = (float*)carve((size_t)N_NODES * 4);
  int*    IDX15 = (int*)carve((size_t)N_NODES * 15 * 4);
  int*    DVC   = (int*)carve((size_t)N_NODES * 4);
  float*  DVINV = (float*)carve((size_t)N_NODES * 4);
  float*  XW1   = (float*)carve((size_t)N_NODES * HID_DIM * 4);
  float*  AG1   = (float*)carve((size_t)N_NODES * HID_DIM * 4);
  __bf16* A1HI  = (__bf16*)carve((size_t)N_NODES * HID_DIM * 2);
  __bf16* A1LO  = (__bf16*)carve((size_t)N_NODES * HID_DIM * 2);
  float*  XW2   = (float*)carve((size_t)N_NODES * FEAT_DIM * 4);
  float*  AG2   = (float*)carve((size_t)N_NODES * FEAT_DIM * 4);
  __bf16* W1THI = (__bf16*)carve((size_t)HID_DIM * C_DIM * 2);
  __bf16* W1TLO = (__bf16*)carve((size_t)HID_DIM * C_DIM * 2);
  __bf16* W2THI = (__bf16*)carve((size_t)FEAT_DIM * HID_DIM * 2);
  __bf16* W2TLO = (__bf16*)carve((size_t)FEAT_DIM * HID_DIM * 2);
  float*  MEANGT = (float*)carve(FEAT_DIM * 4);
  float*  MEAN1  = (float*)carve(FEAT_DIM * 4);
  if (off > ws_size) return;   // insufficient scratch

  float* ZEDGE = DMAT;  // [NEDGE, F] alias inside the 64MB distance buffer

  // Weight transposes (once per call)
  {
    size_t t1 = (size_t)HID_DIM * C_DIM;
    wsplit_kernel<<<(unsigned)((t1 + 255) / 256), 256, 0, stream>>>(W1, W1THI, W1TLO, C_DIM, HID_DIM);
    size_t t2 = (size_t)FEAT_DIM * HID_DIM;
    wsplit_kernel<<<(unsigned)((t2 + 255) / 256), 256, 0, stream>>>(W2, W2THI, W2TLO, HID_DIM, FEAT_DIM);
  }

  const float* xs[2]    = {fts_gt, fts1};
  float*       means[2] = {MEANGT, MEAN1};

  for (int ii = 0; ii < 2; ++ii) {
    const float* X = xs[ii];

    // 1) split to bf16x2 + row norms
    split_norm_kernel<<<N_NODES, 256, 0, stream>>>(X, XHI, XLO, NORMS);

    // 2) pairwise sq-dist Gram via WMMA: D = max(n_i + n_j - 2 X X^T, 0)
    gemm_wmma_bf16x2<1><<<dim3(N_NODES / 64, N_NODES / 128), 256, 0, stream>>>(
        XHI, XLO, XHI, XLO, DMAT, NORMS, N_NODES, N_NODES, C_DIM);

    // 3) exact top-15 per row (serves k=5,10,15 prefixes)
    topk15_kernel<<<N_NODES, 256, 0, stream>>>(DMAT, IDX15);

    // 4) node degrees over concat hypergraph -> dv^{-1/2}
    hipMemsetAsync(DVC, 0, (size_t)N_NODES * 4, stream);
    degcount_kernel<<<(NEDGE + 255) / 256, 256, 0, stream>>>(IDX15, DVC);
    dvinv_kernel<<<(N_NODES + 255) / 256, 256, 0, stream>>>(DVC, DVINV);

    // 5) layer 1:  XW1 = X @ W1  (WMMA), then M-propagation, relu, resplit
    gemm_wmma_bf16x2<0><<<dim3(HID_DIM / 64, N_NODES / 128), 256, 0, stream>>>(
        XHI, XLO, W1THI, W1TLO, XW1, nullptr, N_NODES, HID_DIM, C_DIM);
    gather_kernel<<<NEDGE, 128, 0, stream>>>(IDX15, DVINV, XW1, ZEDGE, HID_DIM);
    hipMemsetAsync(AG1, 0, (size_t)N_NODES * HID_DIM * 4, stream);
    scatter_kernel<<<NEDGE, 128, 0, stream>>>(IDX15, ZEDGE, AG1, HID_DIM);
    {
      size_t tot = (size_t)N_NODES * HID_DIM;
      scale_relu_split_kernel<<<(unsigned)((tot + 255) / 256), 256, 0, stream>>>(
          AG1, DVINV, A1HI, A1LO, HID_DIM);
    }

    // 6) layer 2:  XW2 = h @ W2 (WMMA), M-propagation, dv-scaled column mean
    gemm_wmma_bf16x2<0><<<dim3(FEAT_DIM / 64, N_NODES / 128), 256, 0, stream>>>(
        A1HI, A1LO, W2THI, W2TLO, XW2, nullptr, N_NODES, FEAT_DIM, HID_DIM);
    gather_kernel<<<NEDGE, 128, 0, stream>>>(IDX15, DVINV, XW2, ZEDGE, FEAT_DIM);
    hipMemsetAsync(AG2, 0, (size_t)N_NODES * FEAT_DIM * 4, stream);
    scatter_kernel<<<NEDGE, 128, 0, stream>>>(IDX15, ZEDGE, AG2, FEAT_DIM);
    colmean_kernel<<<FEAT_DIM, 256, 0, stream>>>(AG2, DVINV, means[ii], FEAT_DIM);
  }

  finalize_kernel<<<1, 128, 0, stream>>>(MEANGT, MEAN1, out);
}
